// PSNR_SMIC_58626303590537
// MI455X (gfx1250) — compile-verified, matrix operand order checked by reading, not run on hardware
//
#include <hip/hip_runtime.h>
#include <hip/hip_bf16.h>

typedef _Float16 half_t;
typedef __attribute__((ext_vector_type(8)))  _Float16 v8h;
typedef __attribute__((ext_vector_type(16))) _Float16 v16h;
typedef __attribute__((ext_vector_type(8)))  float    v8f;
typedef __attribute__((ext_vector_type(4)))  int      v4i;

static __device__ __forceinline__ v16h cat8(v8h lo, v8h hi) {
  return __builtin_shufflevector(lo, hi, 0,1,2,3,4,5,6,7,8,9,10,11,12,13,14,15);
}
static __device__ __forceinline__ v8f wmma_f16(v16h a, v16h b, v8f c) {
  // (neg_a, A, neg_b, B, c_mod, C, reuse_a, reuse_b)
  return __builtin_amdgcn_wmma_f32_16x16x32_f16(false, a, false, b, (short)0, c, false, false);
}

// CDNA5 async global->LDS staging (ASYNCcnt) if the toolchain exposes it.
// Probe-learned signature: param 1 is a generic v4i* (the global source);
// LDS destination second; then imm offset, imm cpol.
#if defined(__AMDGCN__) && \
    __has_builtin(__builtin_amdgcn_global_load_async_to_lds_b128) && \
    __has_builtin(__builtin_amdgcn_s_wait_asynccnt)
#define USE_ASYNC_LDS 1
#define ASYNC_COPY_B128(gsrc, ldst) \
  __builtin_amdgcn_global_load_async_to_lds_b128((v4i*)(const void*)(gsrc), \
                                                 (v4i*)(void*)(ldst), 0, 0)
#define ASYNC_WAIT() __builtin_amdgcn_s_wait_asynccnt(0)
#else
#define USE_ASYNC_LDS 0
#define ASYNC_COPY_B128(gsrc, ldst)
#define ASYNC_WAIT()
#endif

// ---------------------------------------------------------------------------
// Layout converters
// ---------------------------------------------------------------------------

// f32 NCHW [3][H][W] -> f16 channel-last [H][W][32] (channels 3..31 zero)
__global__ void img_pad_chlast_k(const float* __restrict__ x, half_t* __restrict__ o,
                                 int H, int W) {
  int i = blockIdx.x * blockDim.x + threadIdx.x;
  int total = H * W * 32;
  if (i >= total) return;
  int c = i & 31;
  int p = i >> 5;
  float v = (c < 3) ? x[c * H * W + p] : 0.0f;
  o[i] = (half_t)v;
}

// OIHW f32 [Cout][Cin][3][3] -> f16 [9][Cout][CinP] (cin padded with zeros)
__global__ void repack_w_k(const float* __restrict__ w, half_t* __restrict__ o,
                           int Cout, int Cin, int CinP) {
  int i = blockIdx.x * blockDim.x + threadIdx.x;
  int total = 9 * Cout * CinP;
  if (i >= total) return;
  int ci = i % CinP;
  int rest = i / CinP;
  int co = rest % Cout;
  int t = rest / Cout;
  float v = (ci < Cin) ? w[((size_t)co * Cin + ci) * 9 + t] : 0.0f;
  o[i] = (half_t)v;
}

__global__ void f32_to_f16_k(const float* __restrict__ a, half_t* __restrict__ o, int n) {
  int i = blockIdx.x * blockDim.x + threadIdx.x;
  if (i < n) o[i] = (half_t)a[i];
}

// ---------------------------------------------------------------------------
// 3x3 conv (SAME, stride 1) + bias + ReLU, channel-last, implicit GEMM WMMA.
// in:  [H][W][Cin]  f16 (Cin % 32 == 0)
// wA:  [9][Cout][Cin] f16
// out: [H][W][Cout] f16 (Cout % 64 == 0)
// block = 128 threads (4 waves); wave w -> 16 couts, 64 pixels (4 x N16 tiles)
// grid = ((W/64)*H, Cout/64)
// Double-buffered LDS input tile; async global->LDS staging when available.
// ---------------------------------------------------------------------------
#define NTILE 64
__global__ __launch_bounds__(128) void conv3x3_wmma_k(
    const half_t* __restrict__ in, const half_t* __restrict__ wA,
    const float* __restrict__ bias, half_t* __restrict__ out,
    int Cin, int Cout, int H, int W) {
  __shared__ half_t lds[2][3 * 66 * 32];  // 2 x [row][px(+halo)][cin] = 24.75 KB

  const int wtiles = W / NTILE;
  const int ox0 = (blockIdx.x % wtiles) * NTILE;
  const int oy = blockIdx.x / wtiles;
  const int wave = threadIdx.x >> 5;
  const int lane = threadIdx.x & 31;
  const int hi = lane >> 4;
  const int n = lane & 15;
  const int co_base = blockIdx.y * 64 + wave * 16;
  const int kblocks = Cin >> 5;

  // Stage one [3][66][32] f16 tile (zero-padded halo) for K-block kb into buf.
  auto stage = [&](int kb, half_t* buf) {
    for (int idx = threadIdx.x; idx < 3 * 66 * 4; idx += 128) {
      int q = idx & 3;
      int px = (idx >> 2) % 66;
      int row = (idx >> 2) / 66;
      int y = oy + row - 1;
      int x = ox0 + px - 1;
      half_t* dst = buf + ((row * 66 + px) * 32) + q * 8;
      if ((unsigned)y < (unsigned)H && (unsigned)x < (unsigned)W) {
        const half_t* src = in + ((size_t)y * W + x) * Cin + (size_t)kb * 32 + q * 8;
#if USE_ASYNC_LDS
        ASYNC_COPY_B128(src, dst);
#else
        *(v8h*)dst = *(const v8h*)src;
#endif
      } else {
        v8h z = {};
        *(v8h*)dst = z;
      }
    }
  };

  v8f acc0 = {}, acc1 = {}, acc2 = {}, acc3 = {};

  stage(0, lds[0]);
  ASYNC_WAIT();
  __syncthreads();

  for (int kb = 0; kb < kblocks; ++kb) {
    if (kb + 1 < kblocks) stage(kb + 1, lds[(kb + 1) & 1]);  // overlap with compute

    const half_t* cur = lds[kb & 1];
    const half_t* wbase = wA + ((size_t)co_base + n) * Cin + (size_t)kb * 32 + hi * 8;
#pragma unroll
    for (int t = 0; t < 9; ++t) {
      const int dy = t / 3, dx = t % 3;
      const half_t* wt = wbase + (size_t)t * Cout * Cin;
      v16h Af = cat8(*(const v8h*)wt, *(const v8h*)(wt + 16));
      v16h Bf[4];
#pragma unroll
      for (int nt = 0; nt < 4; ++nt) {
        const half_t* bp = cur + ((dy * 66 + (nt * 16 + n + dx)) * 32) + hi * 8;
        Bf[nt] = cat8(*(const v8h*)bp, *(const v8h*)(bp + 16));
      }
      // 4 independent WMMAs back-to-back (one DS wait instead of four).
      acc0 = wmma_f16(Af, Bf[0], acc0);
      acc1 = wmma_f16(Af, Bf[1], acc1);
      acc2 = wmma_f16(Af, Bf[2], acc2);
      acc3 = wmma_f16(Af, Bf[3], acc3);
    }
    ASYNC_WAIT();
    __syncthreads();
  }

  // Epilogue: bias + ReLU + f16 store (contiguous 8 couts per lane).
#pragma unroll
  for (int nt = 0; nt < 4; ++nt) {
    v8f acc = (nt == 0) ? acc0 : (nt == 1) ? acc1 : (nt == 2) ? acc2 : acc3;
    int px = ox0 + nt * 16 + n;
    v8h o8;
#pragma unroll
    for (int r = 0; r < 8; ++r) {
      int co = co_base + hi * 8 + r;
      float v = acc[r] + bias[co];
      v = v > 0.0f ? v : 0.0f;
      o8[r] = (half_t)v;
    }
    *(v8h*)(out + ((size_t)oy * W + px) * Cout + co_base + hi * 8) = o8;
  }
}

// ---------------------------------------------------------------------------
// L2 pool: hann 3x3 (normalized), stride 2, pad 1; sqrt(conv(x^2)+1e-12)
// channel-last [H][W][C] -> [H/2][W/2][C]
// ---------------------------------------------------------------------------
__global__ void l2pool_k(const half_t* __restrict__ in, half_t* __restrict__ out,
                         int C, int H, int W) {
  int OH = H >> 1, OW = W >> 1;
  int i = blockIdx.x * blockDim.x + threadIdx.x;
  int total = OH * OW * C;
  if (i >= total) return;
  int c = i % C;
  int p = i / C;
  int ox = p % OW;
  int oy = p / OW;
  const float w1[3] = {0.5f, 1.0f, 0.5f};
  float acc = 0.0f;
#pragma unroll
  for (int r = 0; r < 3; ++r) {
    int y = 2 * oy - 1 + r;
    if ((unsigned)y >= (unsigned)H) continue;
#pragma unroll
    for (int s = 0; s < 3; ++s) {
      int x = 2 * ox - 1 + s;
      if ((unsigned)x >= (unsigned)W) continue;
      float v = (float)in[((size_t)y * W + x) * C + c];
      acc += v * v * (w1[r] * w1[s] * 0.25f);  // / sum(k)=4
    }
  }
  out[i] = (half_t)sqrtf(acc + 1e-12f);
}

// ---------------------------------------------------------------------------
// 1x1 projection conv C->32 via WMMA: D[32, HW] = Wp[32,C] x F[C,HW]
// f:  [HW][C] f16 channel-last;  wp: [32][C] f16;  out: [HW][32] f32
// block 128 (4 waves, 16 px each); grid = HW/64
// ---------------------------------------------------------------------------
__global__ __launch_bounds__(128) void proj_wmma_k(
    const half_t* __restrict__ f, const half_t* __restrict__ wproj,
    float* __restrict__ out, int C, int HW) {
  int wave = threadIdx.x >> 5, lane = threadIdx.x & 31;
  int hi = lane >> 4, n = lane & 15;
  int px0 = blockIdx.x * 64 + wave * 16;
  v8f acc0 = {}, acc1 = {};
  int kblocks = C >> 5;
  for (int kb = 0; kb < kblocks; ++kb) {
    const half_t* bp = f + ((size_t)(px0 + n)) * C + kb * 32 + hi * 8;
    v16h Bf = cat8(*(const v8h*)bp, *(const v8h*)(bp + 16));
    const half_t* a0 = wproj + (size_t)n * C + kb * 32 + hi * 8;
    const half_t* a1 = wproj + (size_t)(16 + n) * C + kb * 32 + hi * 8;
    v16h A0 = cat8(*(const v8h*)a0, *(const v8h*)(a0 + 16));
    v16h A1 = cat8(*(const v8h*)a1, *(const v8h*)(a1 + 16));
    acc0 = wmma_f16(A0, Bf, acc0);
    acc1 = wmma_f16(A1, Bf, acc1);
  }
  int px = px0 + n;
#pragma unroll
  for (int r = 0; r < 8; ++r) {
    out[(size_t)px * 32 + hi * 8 + r] = acc0[r];
    out[(size_t)px * 32 + 16 + hi * 8 + r] = acc1[r];
  }
}

// ---------------------------------------------------------------------------
// MIC surrogate: per (patch, channel) 5x5 joint histogram NMI over 49 samples.
// dp/rp: [Wf*Hf][32] f32; one wave32 block per patch, lane = channel.
// mic[p] = 1 - mean_c(nmi)
// ---------------------------------------------------------------------------
__global__ void mic_k(const float* __restrict__ dp, const float* __restrict__ rp,
                      float* __restrict__ mic, int hp, int wp, int Wf) {
  int p = blockIdx.x;
  int c = threadIdx.x;  // 0..31
  int py = p / wp, px = p % wp;

  float ulo = 3e38f, uhi = -3e38f, vlo = 3e38f, vhi = -3e38f;
  for (int i = 0; i < 7; ++i)
    for (int j = 0; j < 7; ++j) {
      int idx = (((py * 7 + i) * Wf) + px * 7 + j) * 32 + c;
      float u = dp[idx], v = rp[idx];
      ulo = fminf(ulo, u); uhi = fmaxf(uhi, u);
      vlo = fminf(vlo, v); vhi = fmaxf(vhi, v);
    }
  float us = 5.0f / (uhi - ulo + 1e-12f);
  float vs = 5.0f / (vhi - vlo + 1e-12f);

  float hist[25];
#pragma unroll
  for (int b = 0; b < 25; ++b) hist[b] = 0.0f;
  for (int i = 0; i < 7; ++i)
    for (int j = 0; j < 7; ++j) {
      int idx = (((py * 7 + i) * Wf) + px * 7 + j) * 32 + c;
      int iu = (int)floorf((dp[idx] - ulo) * us);
      int iv = (int)floorf((rp[idx] - vlo) * vs);
      iu = iu < 0 ? 0 : (iu > 4 ? 4 : iu);
      iv = iv < 0 ? 0 : (iv > 4 ? 4 : iv);
      int code = iu * 5 + iv;
#pragma unroll
      for (int b = 0; b < 25; ++b) hist[b] += (code == b) ? 1.0f : 0.0f;
    }

  const float inv_n = 1.0f / 49.0f;
  float pxm[5] = {0, 0, 0, 0, 0}, pym[5] = {0, 0, 0, 0, 0};
#pragma unroll
  for (int a = 0; a < 5; ++a)
#pragma unroll
    for (int b = 0; b < 5; ++b) {
      float pj = hist[a * 5 + b] * inv_n;
      pxm[a] += pj;
      pym[b] += pj;
    }
  float mi = 0.0f;
#pragma unroll
  for (int a = 0; a < 5; ++a)
#pragma unroll
    for (int b = 0; b < 5; ++b) {
      float pj = hist[a * 5 + b] * inv_n;
      if (pj > 0.0f) mi += pj * logf(pj / (pxm[a] * pym[b] + 1e-12f) + 1e-12f);
    }
  float val = mi / logf(5.0f);
  for (int off = 16; off; off >>= 1) val += __shfl_xor(val, off, 32);
  if (c == 0) mic[p] = 1.0f - val * (1.0f / 32.0f);
}

// ---------------------------------------------------------------------------
// MSE pieces + resize + reduction
// ---------------------------------------------------------------------------
__global__ void d2_k(const float* __restrict__ x, const float* __restrict__ y,
                     float* __restrict__ d2, int HW) {
  int i = blockIdx.x * blockDim.x + threadIdx.x;
  if (i >= HW) return;
  float s = 0.0f;
#pragma unroll
  for (int c = 0; c < 3; ++c) {
    float d = x[(size_t)c * HW + i] - y[(size_t)c * HW + i];
    s += d * d;
  }
  d2[i] = s * (1.0f / 3.0f);
}

__global__ void msebox_k(const float* __restrict__ d2, float* __restrict__ mse,
                         int H, int W) {
  int OH = H - 6, OW = W - 6;
  int i = blockIdx.x * blockDim.x + threadIdx.x;
  if (i >= OH * OW) return;
  int ox = i % OW, oy = i / OW;
  float s = 0.0f;
  for (int r = 0; r < 7; ++r)
    for (int q = 0; q < 7; ++q) s += d2[(size_t)(oy + r) * W + ox + q];
  mse[i] = s * (1.0f / 49.0f);
}

static __device__ __forceinline__ float bilin(const float* img, int ih, int iw,
                                              float yy, float xx) {
  int y0 = (int)floorf(yy);
  int x0 = (int)floorf(xx);
  int y1 = (y0 + 1 < ih) ? y0 + 1 : ih - 1;
  int x1 = (x0 + 1 < iw) ? x0 + 1 : iw - 1;
  float wy = yy - (float)y0, wx = xx - (float)x0;
  float a = img[y0 * iw + x0], b = img[y0 * iw + x1];
  float c = img[y1 * iw + x0], d = img[y1 * iw + x1];
  float top = a * (1.0f - wx) + b * wx;
  float bot = c * (1.0f - wx) + d * wx;
  return top * (1.0f - wy) + bot * wy;
}

__global__ void micfinal_k(const float* __restrict__ mic3, const float* __restrict__ mic4,
                           float* __restrict__ micf, int h3, int w3, int h4, int w4) {
  int i = blockIdx.x * blockDim.x + threadIdx.x;
  if (i >= h3 * w3) return;
  int ix = i % w3, iy = i / w3;
  float yy = (h3 > 1) ? iy * (float)(h4 - 1) / (float)(h3 - 1) : 0.0f;
  float xx = (w3 > 1) ? ix * (float)(w4 - 1) / (float)(w3 - 1) : 0.0f;
  micf[i] = 0.5f * (mic3[i] + bilin(mic4, h4, w4, yy, xx));
}

__global__ void zero_k(float* p, int n) {
  int i = blockIdx.x * blockDim.x + threadIdx.x;
  if (i < n) p[i] = 0.0f;
}

__global__ void reduce_k(const float* __restrict__ mse, const float* __restrict__ micf,
                         int OH, int OW, int h3, int w3, float* __restrict__ acc) {
  __shared__ float sdata[256];
  int i = blockIdx.x * 256 + threadIdx.x;
  float v = 0.0f;
  if (i < OH * OW) {
    int oy = i / OW, ox = i % OW;
    float yy = oy * (float)(h3 - 1) / (float)(OH - 1);
    float xx = ox * (float)(w3 - 1) / (float)(OW - 1);
    v = mse[i] * bilin(micf, h3, w3, yy, xx);
  }
  sdata[threadIdx.x] = v;
  __syncthreads();
  for (int s = 128; s; s >>= 1) {
    if (threadIdx.x < s) sdata[threadIdx.x] += sdata[threadIdx.x + s];
    __syncthreads();
  }
  if (threadIdx.x == 0) atomicAdd(acc, sdata[0]);
}

__global__ void finalize_k(const float* __restrict__ acc, float* __restrict__ out,
                           float inv_count) {
  float r = acc[0] * inv_count;
  out[0] = 10.0f * log10f(1.0f / (r + 1e-10f));
}

// ---------------------------------------------------------------------------
// Host orchestration
// ---------------------------------------------------------------------------
extern "C" void kernel_launch(void* const* d_in, const int* in_sizes, int n_in,
                              void* d_out, int out_size, void* d_ws, size_t ws_size,
                              hipStream_t stream) {
  const float* X = (const float*)d_in[0];
  const float* Y = (const float*)d_in[1];
  const float* cw[10];
  const float* cb[10];
  for (int i = 0; i < 10; ++i) {
    cw[i] = (const float*)d_in[2 + 2 * i];
    cb[i] = (const float*)d_in[3 + 2 * i];
  }
  const float* pj3 = (const float*)d_in[22];
  const float* pj4 = (const float*)d_in[23];

  // -------- workspace carving --------
  size_t off = 0;
  auto carve = [&](size_t bytes) -> void* {
    void* p = (char*)d_ws + off;
    off += (bytes + 255) & ~(size_t)255;
    return p;
  };
  half_t* actA = (half_t*)carve((size_t)512 * 512 * 64 * 2);
  half_t* actB = (half_t*)carve((size_t)512 * 512 * 64 * 2);
  half_t* f3x = (half_t*)carve((size_t)128 * 128 * 256 * 2);
  half_t* f3y = (half_t*)carve((size_t)128 * 128 * 256 * 2);
  half_t* f4x = (half_t*)carve((size_t)64 * 64 * 512 * 2);
  half_t* f4y = (half_t*)carve((size_t)64 * 64 * 512 * 2);
  half_t* wbuf = (half_t*)carve((size_t)9 * 512 * 512 * 2);
  float* rp3 = (float*)carve((size_t)128 * 128 * 32 * 4);
  float* dp3 = (float*)carve((size_t)128 * 128 * 32 * 4);
  float* rp4 = (float*)carve((size_t)64 * 64 * 32 * 4);
  float* dp4 = (float*)carve((size_t)64 * 64 * 32 * 4);
  float* d2 = (float*)carve((size_t)512 * 512 * 4);
  float* mse = (float*)carve((size_t)506 * 506 * 4);
  float* mic3 = (float*)carve((size_t)18 * 18 * 4);
  float* mic4 = (float*)carve((size_t)9 * 9 * 4);
  float* micf = (float*)carve((size_t)18 * 18 * 4);
  float* accb = (float*)carve(4);

  auto conv_layer = [&](int li, const half_t* in, half_t* outp,
                        int Cin, int CinP, int Cout, int H, int W) {
    int nrep = 9 * Cout * CinP;
    repack_w_k<<<(nrep + 255) / 256, 256, 0, stream>>>(cw[li], wbuf, Cout, Cin, CinP);
    dim3 g((W / NTILE) * H, Cout / 64);
    conv3x3_wmma_k<<<g, 128, 0, stream>>>(in, wbuf, cb[li], outp, CinP, Cout, H, W);
  };
  auto l2pool = [&](const half_t* in, half_t* outp, int C, int H, int W) {
    int n = (H / 2) * (W / 2) * C;
    l2pool_k<<<(n + 255) / 256, 256, 0, stream>>>(in, outp, C, H, W);
  };

  auto run_vgg = [&](const float* img, half_t* f3o, half_t* f4o) {
    int n0 = 512 * 512 * 32;
    img_pad_chlast_k<<<(n0 + 255) / 256, 256, 0, stream>>>(img, actA, 512, 512);
    conv_layer(0, actA, actB, 3, 32, 64, 512, 512);    // conv1_1
    conv_layer(1, actB, actA, 64, 64, 64, 512, 512);   // conv1_2
    l2pool(actA, actB, 64, 512, 512);                  // -> 256x256x64
    conv_layer(2, actB, actA, 64, 64, 128, 256, 256);  // conv2_1
    conv_layer(3, actA, actB, 128, 128, 128, 256, 256);// conv2_2
    l2pool(actB, actA, 128, 256, 256);                 // -> 128x128x128
    conv_layer(4, actA, actB, 128, 128, 256, 128, 128);// conv3_1
    conv_layer(5, actB, actA, 256, 256, 256, 128, 128);// conv3_2
    conv_layer(6, actA, f3o, 256, 256, 256, 128, 128); // conv3_3 (persist)
    l2pool(f3o, actA, 256, 128, 128);                  // -> 64x64x256
    conv_layer(7, actA, actB, 256, 256, 512, 64, 64);  // conv4_1
    conv_layer(8, actB, actA, 512, 512, 512, 64, 64);  // conv4_2
    conv_layer(9, actA, f4o, 512, 512, 512, 64, 64);   // conv4_3 (persist)
  };

  run_vgg(X, f3x, f4x);  // reference features
  run_vgg(Y, f3y, f4y);  // distorted features

  // ---- 1x1 projections (WMMA) ----
  {
    int n3 = 32 * 256;
    f32_to_f16_k<<<(n3 + 255) / 256, 256, 0, stream>>>(pj3, wbuf, n3);
    int HW3 = 128 * 128;
    proj_wmma_k<<<HW3 / 64, 128, 0, stream>>>(f3x, wbuf, rp3, 256, HW3);
    proj_wmma_k<<<HW3 / 64, 128, 0, stream>>>(f3y, wbuf, dp3, 256, HW3);
    int n4 = 32 * 512;
    f32_to_f16_k<<<(n4 + 255) / 256, 256, 0, stream>>>(pj4, wbuf, n4);
    int HW4 = 64 * 64;
    proj_wmma_k<<<HW4 / 64, 128, 0, stream>>>(f4x, wbuf, rp4, 512, HW4);
    proj_wmma_k<<<HW4 / 64, 128, 0, stream>>>(f4y, wbuf, dp4, 512, HW4);
  }

  // ---- MIC maps ----
  mic_k<<<18 * 18, 32, 0, stream>>>(dp3, rp3, mic3, 18, 18, 128);
  mic_k<<<9 * 9, 32, 0, stream>>>(dp4, rp4, mic4, 9, 9, 64);
  micfinal_k<<<(18 * 18 + 255) / 256, 256, 0, stream>>>(mic3, mic4, micf, 18, 18, 9, 9);

  // ---- MSE map ----
  int HW = 512 * 512;
  d2_k<<<(HW + 255) / 256, 256, 0, stream>>>(X, Y, d2, HW);
  int OM = 506 * 506;
  msebox_k<<<(OM + 255) / 256, 256, 0, stream>>>(d2, mse, 512, 512);

  // ---- final weighted mean + PSNR ----
  zero_k<<<1, 32, 0, stream>>>(accb, 1);
  reduce_k<<<(OM + 255) / 256, 256, 0, stream>>>(mse, micf, 506, 506, 18, 18, accb);
  finalize_k<<<1, 1, 0, stream>>>(accb, (float*)d_out, 1.0f / (float)OM);
}
